// GAT_83940840833064
// MI455X (gfx1250) — compile-verified, matrix operand order checked by reading, not run on hardware
//
#include <hip/hip_runtime.h>
#include <hip/hip_bf16.h>
#include <math.h>

typedef __attribute__((ext_vector_type(8)))  float   v8f;
typedef __attribute__((ext_vector_type(16))) __bf16  v16bf;
typedef __attribute__((ext_vector_type(8)))  __bf16  v8bf;

union BF16x16 { v16bf v; v8bf h[2]; };

#define N_NODES 50000
#define N_EDGES 800000
#define ETOT    (N_EDGES + N_NODES)
#define HID     64

// ---------- helpers ----------
static __device__ __forceinline__ unsigned f2ord(float f) {
  unsigned u = __float_as_uint(f);
  return (u & 0x80000000u) ? ~u : (u | 0x80000000u);   // monotone float->uint
}
static __device__ __forceinline__ float ord2f(unsigned u) {
  return (u & 0x80000000u) ? __uint_as_float(u & 0x7fffffffu) : __uint_as_float(~u);
}
static __device__ __forceinline__ void edge_sd(const int* __restrict__ ei, int e,
                                               int& s, int& d) {
  if (e < N_EDGES) { s = ei[e]; d = ei[N_EDGES + e]; }
  else             { s = d = e - N_EDGES; }             // self loops appended
}
static __device__ __forceinline__ float lrelu(float v) {
  return v > 0.f ? v : 0.2f * v;
}

// ---------- utility kernels ----------
__global__ void k_zero_u32(unsigned* __restrict__ p, int n) {
  int i = blockIdx.x * blockDim.x + threadIdx.x;
  if (i < n) p[i] = 0u;
}

__global__ void k_f32_to_bf16(const float* __restrict__ in, __bf16* __restrict__ out, int n) {
  int i = blockIdx.x * blockDim.x + threadIdx.x;
  if (i < n) out[i] = (__bf16)in[i];
}

// W [K, Ncol] fp32 row-major  ->  WT [Ncol, K] bf16 (contiguous in K)
__global__ void k_transpose_bf16(const float* __restrict__ w, __bf16* __restrict__ wt,
                                 int K, int Ncol) {
  int i = blockIdx.x * blockDim.x + threadIdx.x;
  if (i >= K * Ncol) return;
  int k = i / Ncol, n = i % Ncol;
  wt[(size_t)n * K + k] = (__bf16)w[i];
}

// ---------- WMMA GEMM: C[M,64] = A[M,K](bf16) * B[K,64]; BT is [64,K] bf16 ----------
__global__ void k_gemm_bf16_wmma(const __bf16* __restrict__ A,
                                 const __bf16* __restrict__ BT,
                                 float* __restrict__ C, int M, int K) {
  const int wave = threadIdx.x >> 5;
  const int tile = blockIdx.x * (blockDim.x >> 5) + wave;   // 16-row strip
  if (tile * 16 >= M) return;                                // wave-uniform guard
  const int lane = threadIdx.x & 31;
  const int half = lane >> 4;          // K-half selector (A/B), row-half (C)
  const int mrow = lane & 15;          // A row / B column within tile

  v8f acc[4] = {};                     // 4 column tiles of 16 -> 64 output cols

  const __bf16* arow = A + (size_t)(tile * 16 + mrow) * K + 8 * half;
  for (int k0 = 0; k0 < K; k0 += 32) {
    BF16x16 a;
    a.h[0] = *(const v8bf*)(arow + k0);        // K = k0 + 8*half + 0..7
    a.h[1] = *(const v8bf*)(arow + k0 + 16);   // K = k0 + 16 + 8*half + 0..7
#pragma unroll
    for (int nt = 0; nt < 4; nt++) {
      const __bf16* brow = BT + (size_t)(nt * 16 + mrow) * K + 8 * half;
      BF16x16 b;
      b.h[0] = *(const v8bf*)(brow + k0);
      b.h[1] = *(const v8bf*)(brow + k0 + 16);
      acc[nt] = __builtin_amdgcn_wmma_f32_16x16x32_bf16(
          false, a.v, false, b.v, (short)0, acc[nt], false, false);
    }
  }
#pragma unroll
  for (int nt = 0; nt < 4; nt++)
#pragma unroll
    for (int r = 0; r < 8; r++)
      C[(size_t)(tile * 16 + r + 8 * half) * HID + nt * 16 + mrow] = acc[nt][r];
}

// ---------- attention coefficients: a_s/a_d[n,h] = <h[n,h,:], att[h,:]> ----------
__global__ void k_attn(const float* __restrict__ hbuf, const float* __restrict__ att_s,
                       const float* __restrict__ att_d, float* __restrict__ as_,
                       float* __restrict__ ad_, int N, int H, int C) {
  int i = blockIdx.x * blockDim.x + threadIdx.x;
  if (i >= N * H) return;
  int n = i / H, hh = i % H;
  const float* row = hbuf + (size_t)n * H * C + hh * C;
  float s = 0.f, d = 0.f;
  for (int c = 0; c < C; c++) {
    float v = row[c];
    s += v * att_s[hh * C + c];
    d += v * att_d[hh * C + c];
  }
  as_[i] = s; ad_[i] = d;
}

// ---------- edge passes (segment softmax + aggregation) ----------
__global__ void k_edge_max(const int* __restrict__ ei, const float* __restrict__ as_,
                           const float* __restrict__ ad_, unsigned* __restrict__ mx, int H) {
  int i = blockIdx.x * blockDim.x + threadIdx.x;
  if (i >= ETOT * H) return;
  int e = i / H, h = i % H, s, d;
  edge_sd(ei, e, s, d);
  float v = lrelu(as_[s * H + h] + ad_[d * H + h]);
  atomicMax(&mx[d * H + h], f2ord(v));
}

__global__ void k_edge_sum(const int* __restrict__ ei, const float* __restrict__ as_,
                           const float* __restrict__ ad_, const unsigned* __restrict__ mx,
                           float* __restrict__ sm, int H) {
  int i = blockIdx.x * blockDim.x + threadIdx.x;
  if (i >= ETOT * H) return;
  int e = i / H, h = i % H, s, d;
  edge_sd(ei, e, s, d);
  float v = lrelu(as_[s * H + h] + ad_[d * H + h]);
  atomicAdd(&sm[d * H + h], __expf(v - ord2f(mx[d * H + h])));
}

__global__ void k_edge_aggr(const int* __restrict__ ei, const float* __restrict__ as_,
                            const float* __restrict__ ad_, const unsigned* __restrict__ mx,
                            const float* __restrict__ sm, const float* __restrict__ hsrc,
                            float* __restrict__ outp, int H, int C) {
  int i = blockIdx.x * blockDim.x + threadIdx.x;
  if (i >= ETOT * H) return;
  int e = i / H, h = i % H, s, d;
  edge_sd(ei, e, s, d);
  float v = lrelu(as_[s * H + h] + ad_[d * H + h]);
  float alpha = __expf(v - ord2f(mx[d * H + h])) / (sm[d * H + h] + 1e-16f);
  const float* src_row = hsrc + (size_t)s * H * C + h * C;
  float* dst_row = outp + (size_t)d * H * C + h * C;
  __builtin_prefetch(src_row, 0, 1);
  for (int c = 0; c < C; c++)
    atomicAdd(&dst_row[c], alpha * src_row[c]);
}

// ---------- layer finalizers ----------
// h1act = bf16( elu(agg + b1) )
__global__ void k_finalize1(const float* __restrict__ agg, const float* __restrict__ b1,
                            __bf16* __restrict__ h1act, int n) {
  int i = blockIdx.x * blockDim.x + threadIdx.x;
  if (i >= n) return;
  float v = agg[i] + b1[i % HID];
  v = v > 0.f ? v : (__expf(v) - 1.f);   // ELU, alpha=1
  h1act[i] = (__bf16)v;
}

__global__ void k_bias_add(float* __restrict__ outp, const float* __restrict__ b, int n) {
  int i = blockIdx.x * blockDim.x + threadIdx.x;
  if (i >= n) return;
  outp[i] += b[i % HID];
}

// ---------- launcher ----------
extern "C" void kernel_launch(void* const* d_in, const int* in_sizes, int n_in,
                              void* d_out, int out_size, void* d_ws, size_t ws_size,
                              hipStream_t stream) {
  (void)in_sizes; (void)n_in; (void)out_size; (void)ws_size;
  const float* x    = (const float*)d_in[0];
  const int*   ei   = (const int*)  d_in[1];
  const float* W1   = (const float*)d_in[2];
  const float* asw1 = (const float*)d_in[3];
  const float* adw1 = (const float*)d_in[4];
  const float* b1   = (const float*)d_in[5];
  const float* W2   = (const float*)d_in[6];
  const float* asw2 = (const float*)d_in[7];
  const float* adw2 = (const float*)d_in[8];
  const float* b2   = (const float*)d_in[9];
  float* out = (float*)d_out;

  char* wsp = (char*)d_ws;
  size_t off = 0;
  auto alloc = [&](size_t bytes) -> char* {
    char* p = wsp + off;
    off = (off + bytes + 255) & ~(size_t)255;
    return p;
  };
  __bf16* xb    = (__bf16*)  alloc((size_t)N_NODES * 128 * 2);  // also reused as agg1
  __bf16* w1t   = (__bf16*)  alloc(64 * 128 * 2);
  __bf16* w2t   = (__bf16*)  alloc(64 * 64 * 2);
  float*  h1    = (float*)   alloc((size_t)N_NODES * HID * 4);  // also reused as h2
  float*  as1   = (float*)   alloc((size_t)N_NODES * 8 * 4);    // reused as as2
  float*  ad1   = (float*)   alloc((size_t)N_NODES * 8 * 4);    // reused as ad2
  unsigned* m1  = (unsigned*)alloc((size_t)N_NODES * 8 * 4);    // reused as m2
  float*  s1    = (float*)   alloc((size_t)N_NODES * 8 * 4);    // reused as s2
  __bf16* h1act = (__bf16*)  alloc((size_t)N_NODES * HID * 2);

  float* agg1 = (float*)xb;   // xb is dead after gemm1
  float* h2   = h1;           // h1 is dead after finalize1

  const int T = 256;
  auto blk = [](long long n, int t) { return (unsigned)((n + t - 1) / t); };
  const int tiles = N_NODES / 16;                 // 3125, exact
  const unsigned gemm_grid = blk(tiles, T / 32);  // 8 waves per block

  // ---- prep: bf16 activations, transposed bf16 weights
  k_f32_to_bf16   <<<blk((long long)N_NODES * 128, T), T, 0, stream>>>(x, xb, N_NODES * 128);
  k_transpose_bf16<<<blk(128 * 64, T), T, 0, stream>>>(W1, w1t, 128, 64);
  k_transpose_bf16<<<blk(64 * 64, T), T, 0, stream>>>(W2, w2t, 64, 64);

  // ---- layer 1: h1 = x @ W1 (WMMA bf16)
  k_gemm_bf16_wmma<<<gemm_grid, T, 0, stream>>>(xb, w1t, h1, N_NODES, 128);
  k_attn<<<blk((long long)N_NODES * 8, T), T, 0, stream>>>(h1, asw1, adw1, as1, ad1, N_NODES, 8, 8);
  k_zero_u32<<<blk((long long)N_NODES * HID, T), T, 0, stream>>>((unsigned*)agg1, N_NODES * HID);
  k_zero_u32<<<blk((long long)N_NODES * 8, T), T, 0, stream>>>(m1, N_NODES * 8);
  k_zero_u32<<<blk((long long)N_NODES * 8, T), T, 0, stream>>>((unsigned*)s1, N_NODES * 8);
  k_edge_max <<<blk((long long)ETOT * 8, T), T, 0, stream>>>(ei, as1, ad1, m1, 8);
  k_edge_sum <<<blk((long long)ETOT * 8, T), T, 0, stream>>>(ei, as1, ad1, m1, s1, 8);
  k_edge_aggr<<<blk((long long)ETOT * 8, T), T, 0, stream>>>(ei, as1, ad1, m1, s1, h1, agg1, 8, 8);
  k_finalize1<<<blk((long long)N_NODES * HID, T), T, 0, stream>>>(agg1, b1, h1act, N_NODES * HID);

  // ---- layer 2: h2 = elu(h1) @ W2 (WMMA bf16), single head, C=64
  k_gemm_bf16_wmma<<<gemm_grid, T, 0, stream>>>(h1act, w2t, h2, N_NODES, 64);
  k_attn<<<blk(N_NODES, T), T, 0, stream>>>(h2, asw2, adw2, as1, ad1, N_NODES, 1, 64);
  k_zero_u32<<<blk((long long)N_NODES * HID, T), T, 0, stream>>>((unsigned*)out, N_NODES * HID);
  k_zero_u32<<<blk(N_NODES, T), T, 0, stream>>>(m1, N_NODES);
  k_zero_u32<<<blk(N_NODES, T), T, 0, stream>>>((unsigned*)s1, N_NODES);
  k_edge_max <<<blk(ETOT, T), T, 0, stream>>>(ei, as1, ad1, m1, 1);
  k_edge_sum <<<blk(ETOT, T), T, 0, stream>>>(ei, as1, ad1, m1, s1, 1);
  k_edge_aggr<<<blk(ETOT, T), T, 0, stream>>>(ei, as1, ad1, m1, s1, h2, out, 1, 64);
  k_bias_add <<<blk((long long)N_NODES * HID, T), T, 0, stream>>>(out, b2, N_NODES * HID);
}
